// LogicityVisReasoningEngine_8624294330845
// MI455X (gfx1250) — matrix operand order repaired and checked by reading, not compiled, not run on hardware
//
#include <hip/hip_runtime.h>
#include <hip/hip_bf16.h>

typedef __attribute__((ext_vector_type(16))) _Float16 v16h;
typedef __attribute__((ext_vector_type(8)))  float    v8f;
typedef __attribute__((ext_vector_type(2)))  _Float16 h2;

#define NN 200
#define EE (NN*(NN-1))   // 39800 directed edges

enum { ACT_NONE = 0, ACT_RELU = 1, ACT_SIG = 2 };

#define CEILDIV(a,b) (((a)+(b)-1)/(b))

// ---------------------------------------------------------------------------
// f16 WMMA GEMM:  C[m,n] = act( sum_k A[m,k]*Bt[n,k] + bias[n] )
//   A  : [M,K] f16 row-major (K multiple of 32; OOB rows clamped on load)
//   Bt : [Npad,K] f16 row-major, Npad multiple of 64 (pre-transposed + padded)
//   One wave per 16x64 C tile: the A fragment is reused across 4 B fragments,
//   giving 4 back-to-back v_wmma per K-step (XDL operand-reuse friendly).
//   8 waves per block; wave-uniform early-out keeps EXEC all-ones at WMMA.
// ---------------------------------------------------------------------------
template<int ACT, bool SF32, bool SF16>
__global__ __launch_bounds__(256)
void wmma_gemm(const _Float16* __restrict__ A, const _Float16* __restrict__ Bt,
               const float* __restrict__ bias,
               float* __restrict__ Cf, int ldcf,
               _Float16* __restrict__ Ch, int ldch,
               int M, int K, int Nstore)
{
  const int wave   = threadIdx.x >> 5;
  const int lane   = threadIdx.x & 31;
  const int Mtiles = (M + 15) >> 4;
  const int tileM  = blockIdx.x * 8 + wave;    // wave-uniform
  const int tileN  = blockIdx.y;               // 64-wide N tile
  if (tileM >= Mtiles) return;                 // whole wave exits together
  const int l15 = lane & 15;
  const int hi  = lane >> 4;

  int mA = tileM * 16 + l15; if (mA > M - 1) mA = M - 1;   // clamp tail rows
  const _Float16* Arow = A + (size_t)mA * K;
  const _Float16* B0 = Bt + (size_t)(tileN * 64 +  0 + l15) * K;
  const _Float16* B1 = Bt + (size_t)(tileN * 64 + 16 + l15) * K;
  const _Float16* B2 = Bt + (size_t)(tileN * 64 + 32 + l15) * K;
  const _Float16* B3 = Bt + (size_t)(tileN * 64 + 48 + l15) * K;

  v8f acc0 = {}, acc1 = {}, acc2 = {}, acc3 = {};
  for (int k0 = 0; k0 < K; k0 += 32) {
    v16h a, b0, b1, b2, b3;
#pragma unroll
    for (int v = 0; v < 8; ++v) {
      // A 16x32 f16 layout: K=2v + 16*hi (same order as B)
      const int kA = 2*v + 16*hi;
      // B (N x K, A-transposed roles): K=2v + 16*hi
      const int kB = 2*v + 16*hi;
      h2 pa  = *(const h2*)(Arow + k0 + kA);
      h2 pb0 = *(const h2*)(B0 + k0 + kB);
      h2 pb1 = *(const h2*)(B1 + k0 + kB);
      h2 pb2 = *(const h2*)(B2 + k0 + kB);
      h2 pb3 = *(const h2*)(B3 + k0 + kB);
      a [2*v] = pa [0]; a [2*v+1] = pa [1];
      b0[2*v] = pb0[0]; b0[2*v+1] = pb0[1];
      b1[2*v] = pb1[0]; b1[2*v+1] = pb1[1];
      b2[2*v] = pb2[0]; b2[2*v+1] = pb2[1];
      b3[2*v] = pb3[0]; b3[2*v+1] = pb3[1];
    }
    acc0 = __builtin_amdgcn_wmma_f32_16x16x32_f16(false, a, false, b0, (short)0, acc0, false, false);
    acc1 = __builtin_amdgcn_wmma_f32_16x16x32_f16(false, a, false, b1, (short)0, acc1, false, false);
    acc2 = __builtin_amdgcn_wmma_f32_16x16x32_f16(false, a, false, b2, (short)0, acc2, false, false);
    acc3 = __builtin_amdgcn_wmma_f32_16x16x32_f16(false, a, false, b3, (short)0, acc3, false, false);
  }

#pragma unroll
  for (int t = 0; t < 4; ++t) {
    const v8f accv = (t == 0) ? acc0 : (t == 1) ? acc1 : (t == 2) ? acc2 : acc3;
    const int n = tileN * 64 + t * 16 + l15;
    float bv = 0.f;
    if (bias != nullptr && n < Nstore) bv = bias[n];
#pragma unroll
    for (int r = 0; r < 8; ++r) {
      const int m = tileM * 16 + 8*hi + r;     // C layout: VGPR r -> row r + 8*hi
      if (m < M && n < Nstore) {
        float x = accv[r] + bv;
        if (ACT == ACT_RELU)     x = fmaxf(x, 0.f);
        else if (ACT == ACT_SIG) x = 1.f / (1.f + __expf(-x));
        if (SF32) Cf[(size_t)m * ldcf + n] = x;
        if (SF16) Ch[(size_t)m * ldch + n] = (_Float16)x;
      }
    }
  }
}

// ---- weight prep: Wt[n*Kpad+k] = (k<Kin && n<Nin) ? W[k*Nin+n] : 0 ----------
__global__ void transpose_pad(const float* __restrict__ W, _Float16* __restrict__ Wt,
                              int Kin, int Nin, int Kpad, int Npad)
{
  int idx = blockIdx.x * blockDim.x + threadIdx.x;
  if (idx >= Kpad * Npad) return;
  int k = idx % Kpad, n = idx / Kpad;
  float v = (k < Kin && n < Nin) ? W[(size_t)k * Nin + n] : 0.f;
  Wt[idx] = (_Float16)v;
}

// wp2 [128,4096] -> Wt[(h*4+o)*1024 + i] = wp2[h*4096 + i*4 + o]  (f16, [512,1024])
__global__ void permute_wp2(const float* __restrict__ wp2, _Float16* __restrict__ Wt)
{
  int idx = blockIdx.x * blockDim.x + threadIdx.x;
  if (idx >= 512 * 1024) return;
  int i = idx & 1023, r = idx >> 10;
  Wt[idx] = (_Float16)wp2[(size_t)(r >> 2) * 4096 + i * 4 + (r & 3)];
}

__global__ void f32_to_f16(const float* __restrict__ in, _Float16* __restrict__ out, int n)
{
  for (int i = blockIdx.x * blockDim.x + threadIdx.x; i < n; i += gridDim.x * blockDim.x)
    out[i] = (_Float16)in[i];
}

// pair features [E,32] f16 (cols 16..31 zero-pad); also writes higher_pri to out col 3
__global__ void build_pair(const float* __restrict__ bbox, const float* __restrict__ dirs,
                           const float* __restrict__ pri,
                           _Float16* __restrict__ pairh, float* __restrict__ edge_out)
{
  int e = blockIdx.x * blockDim.x + threadIdx.x;
  if (e >= EE) return;
  int src = e / (NN - 1);
  int j   = e - src * (NN - 1);
  int dst = j + (j >= src ? 1 : 0);
  _Float16* p = pairh + (size_t)e * 32;
  const float inv = 1.0f / 1024.0f;
#pragma unroll
  for (int c = 0; c < 4; ++c) {
    p[c]      = (_Float16)(bbox[src*4 + c] * inv);
    p[4 + c]  = (_Float16)(dirs[src*4 + c]);
    p[8 + c]  = (_Float16)(bbox[dst*4 + c] * inv);
    p[12 + c] = (_Float16)(dirs[dst*4 + c]);
  }
#pragma unroll
  for (int c = 16; c < 32; ++c) p[c] = (_Float16)0.f;
  edge_out[(size_t)e * 4 + 3] = (pri[src] > pri[dst]) ? 1.f : 0.f;
}

// node_concepts_explicit[0] = sigmoid(NC @ wi + bi)   (200x4)
__global__ void node_explicit(const float* __restrict__ NC, const float* __restrict__ wi,
                              const float* __restrict__ bi, float* __restrict__ out)
{
  int idx = blockIdx.x * blockDim.x + threadIdx.x;
  if (idx >= NN * 4) return;
  int o = idx & 3, n = idx >> 2;
  const float* x = NC + (size_t)n * 1024;
  float acc = bi[o];
  for (int i = 0; i < 1024; ++i) acc += x[i] * wi[i * 4 + o];
  out[idx] = 1.f / (1.f + __expf(-acc));
}

// ehp[e,h] = relu(sum_c ea[e,c]*wp1[c,h] + bp1[h])   (E x 128)
__global__ void edge_hidden(const float* __restrict__ ea, const float* __restrict__ wp1,
                            const float* __restrict__ bp1, float* __restrict__ ehp)
{
  int idx = blockIdx.x * blockDim.x + threadIdx.x;
  if (idx >= EE * 128) return;
  int h = idx & 127, e = idx >> 7;
  const float* a = ea + (size_t)e * 4;
  float acc = bp1[h] + a[0]*wp1[h] + a[1]*wp1[128 + h] + a[2]*wp1[256 + h] + a[3]*wp1[384 + h];
  ehp[idx] = fmaxf(acc, 0.f);
}

// xb[n,o] = NC[n]·bp2[:,o] ; rootterm[n,o] = NC[n]·root_w[:,o] + root_b[o]
__global__ void node_proj(const float* __restrict__ NC, const float* __restrict__ bp2,
                          const float* __restrict__ rootw, const float* __restrict__ rootb,
                          float* __restrict__ xb, float* __restrict__ rootterm)
{
  int idx = blockIdx.x * blockDim.x + threadIdx.x;
  if (idx >= NN * 4) return;
  int o = idx & 3, n = idx >> 2;
  const float* x = NC + (size_t)n * 1024;
  float a1 = 0.f, a2 = rootb[o];
  for (int i = 0; i < 1024; ++i) {
    a1 += x[i] * bp2[i * 4 + o];
    a2 += x[i] * rootw[i * 4 + o];
  }
  xb[idx] = a1; rootterm[idx] = a2;
}

// msg[e,o] = sum_h ehp[e,h]*P[src, h*4+o] + xb[src,o]
__global__ __launch_bounds__(256)
void edge_msg(const float* __restrict__ ehp, const float* __restrict__ P,
              const float* __restrict__ xb, float* __restrict__ msg)
{
  int e = blockIdx.x * blockDim.x + threadIdx.x;
  if (e >= EE) return;
  int src = e / (NN - 1);
  const float* pr = P + (size_t)src * 512;
  const float* eh = ehp + (size_t)e * 128;
  float a0 = xb[src*4+0], a1 = xb[src*4+1], a2 = xb[src*4+2], a3 = xb[src*4+3];
  for (int h = 0; h < 128; ++h) {
    float w = eh[h];
    const float4 pv = *(const float4*)(pr + h * 4);
    a0 += w * pv.x; a1 += w * pv.y; a2 += w * pv.z; a3 += w * pv.w;
  }
  float* m = msg + (size_t)e * 4;
  m[0] = a0; m[1] = a1; m[2] = a2; m[3] = a3;
}

// agg[n,o] = sum over incoming edges (deterministic tree reduce) + rootterm
__global__ __launch_bounds__(256)
void aggregate(const float* __restrict__ msg, const float* __restrict__ rootterm,
               float* __restrict__ out)
{
  __shared__ float red[4][256];
  int n = blockIdx.x, t = threadIdx.x;
  float a0=0.f, a1=0.f, a2=0.f, a3=0.f;
  for (int s = t; s < NN; s += 256) {
    if (s == n) continue;
    int e = s * (NN - 1) + n - (n > s ? 1 : 0);
    const float* m = msg + (size_t)e * 4;
    a0 += m[0]; a1 += m[1]; a2 += m[2]; a3 += m[3];
  }
  red[0][t]=a0; red[1][t]=a1; red[2][t]=a2; red[3][t]=a3;
  __syncthreads();
  for (int st = 128; st > 0; st >>= 1) {
    if (t < st) {
      red[0][t]+=red[0][t+st]; red[1][t]+=red[1][t+st];
      red[2][t]+=red[2][t+st]; red[3][t]+=red[3][t+st];
    }
    __syncthreads();
  }
  if (t < 4) out[n * 4 + t] = red[t][0] + rootterm[n * 4 + t];
}

// ---------------------------------------------------------------------------
extern "C" void kernel_launch(void* const* d_in, const int* in_sizes, int n_in,
                              void* d_out, int out_size, void* d_ws, size_t ws_size,
                              hipStream_t stream)
{
  (void)in_sizes; (void)n_in; (void)out_size; (void)ws_size;
  const float* roi  = (const float*)d_in[0];   // [2,200,2048] -> use batch 0
  const float* bbox = (const float*)d_in[1];   // [2,200,4]
  const float* dirs = (const float*)d_in[2];
  const float* pri  = (const float*)d_in[3];   // [2,200]
  const float* w1  = (const float*)d_in[4];  const float* b1  = (const float*)d_in[5];
  const float* w2  = (const float*)d_in[6];  const float* b2  = (const float*)d_in[7];
  const float* w3  = (const float*)d_in[8];  const float* b3  = (const float*)d_in[9];
  const float* wi  = (const float*)d_in[10]; const float* bi  = (const float*)d_in[11];
  const float* we1 = (const float*)d_in[12]; const float* be1 = (const float*)d_in[13];
  const float* we2 = (const float*)d_in[14]; const float* be2 = (const float*)d_in[15];
  const float* we3 = (const float*)d_in[16]; const float* be3 = (const float*)d_in[17];
  const float* wp1 = (const float*)d_in[18]; const float* bp1 = (const float*)d_in[19];
  const float* wp2 = (const float*)d_in[20]; const float* bp2 = (const float*)d_in[21];
  const float* rootw = (const float*)d_in[22]; const float* rootb = (const float*)d_in[23];

  float* out_actions = (float*)d_out;          // [200,4]
  float* out_expl    = (float*)d_out + 800;    // [200,4]
  float* out_edge    = (float*)d_out + 1600;   // [E,4]

  char* ws = (char*)d_ws; size_t off = 0;
  auto alloc = [&](size_t bytes) -> void* {
    off = (off + 255) & ~(size_t)255;
    void* p = ws + off; off += bytes; return p;
  };
  _Float16* w1t   = (_Float16*)alloc((size_t)512*2048*2);
  _Float16* w2t   = (_Float16*)alloc((size_t)256*512*2);
  _Float16* w3t   = (_Float16*)alloc((size_t)1024*256*2);
  _Float16* we1t  = (_Float16*)alloc((size_t)256*32*2);
  _Float16* we2t  = (_Float16*)alloc((size_t)64*256*2);
  _Float16* we3t  = (_Float16*)alloc((size_t)64*64*2);   // Npad=64 (zero rows 3..63)
  _Float16* wp2p  = (_Float16*)alloc((size_t)512*1024*2);
  _Float16* Xh    = (_Float16*)alloc((size_t)NN*2048*2);
  _Float16* H1h   = (_Float16*)alloc((size_t)NN*512*2);
  _Float16* H2h   = (_Float16*)alloc((size_t)NN*256*2);
  _Float16* NCh   = (_Float16*)alloc((size_t)NN*1024*2);
  float*    NCf   = (float*)   alloc((size_t)NN*1024*4);
  _Float16* pairh = (_Float16*)alloc((size_t)EE*32*2);
  _Float16* EH1h  = (_Float16*)alloc((size_t)EE*256*2);
  _Float16* EH2h  = (_Float16*)alloc((size_t)EE*64*2);
  float*    ehp   = (float*)   alloc((size_t)EE*128*4);
  float*    P     = (float*)   alloc((size_t)NN*512*4);
  float*    xb    = (float*)   alloc((size_t)NN*4*4);
  float*    rtm   = (float*)   alloc((size_t)NN*4*4);
  float*    msg   = (float*)   alloc((size_t)EE*4*4);

  // ---- weight prep (f16 transposed, zero padded) ----
  transpose_pad<<<CEILDIV(2048*512,256),256,0,stream>>>(w1,  w1t, 2048,512, 2048,512);
  transpose_pad<<<CEILDIV(512*256,256), 256,0,stream>>>(w2,  w2t, 512,256,  512,256);
  transpose_pad<<<CEILDIV(256*1024,256),256,0,stream>>>(w3,  w3t, 256,1024, 256,1024);
  transpose_pad<<<CEILDIV(32*256,256),  256,0,stream>>>(we1, we1t, 16,256,  32,256);
  transpose_pad<<<CEILDIV(256*64,256),  256,0,stream>>>(we2, we2t, 256,64,  256,64);
  transpose_pad<<<CEILDIV(64*64,256),   256,0,stream>>>(we3, we3t, 64,3,    64,64);
  permute_wp2  <<<CEILDIV(512*1024,256),256,0,stream>>>(wp2, wp2p);

  // ---- node pipeline (batch 0 only) ----
  f32_to_f16<<<1024,256,0,stream>>>(roi, Xh, NN*2048);
  wmma_gemm<ACT_RELU,false,true><<<dim3(2,8),256,0,stream>>>
      (Xh, w1t, b1, nullptr,0, H1h,512, NN,2048,512);
  wmma_gemm<ACT_RELU,false,true><<<dim3(2,4),256,0,stream>>>
      (H1h, w2t, b2, nullptr,0, H2h,256, NN,512,256);
  wmma_gemm<ACT_NONE,true,true><<<dim3(2,16),256,0,stream>>>
      (H2h, w3t, b3, NCf,1024, NCh,1024, NN,256,1024);
  node_explicit<<<CEILDIV(NN*4,256),256,0,stream>>>(NCf, wi, bi, out_expl);

  // ---- edge pipeline ----
  build_pair<<<CEILDIV(EE,256),256,0,stream>>>(bbox, dirs, pri, pairh, out_edge);
  wmma_gemm<ACT_RELU,false,true><<<dim3(311,4),256,0,stream>>>
      (pairh, we1t, be1, nullptr,0, EH1h,256, EE,32,256);
  wmma_gemm<ACT_RELU,false,true><<<dim3(311,1),256,0,stream>>>
      (EH1h, we2t, be2, nullptr,0, EH2h,64, EE,256,64);
  wmma_gemm<ACT_SIG,true,false><<<dim3(311,1),256,0,stream>>>
      (EH2h, we3t, be3, out_edge,4, nullptr,0, EE,64,3);

  // ---- NNConv, reassociated: P = NC @ perm(wp2)  (200x1024x512 GEMM) ----
  edge_hidden<<<CEILDIV(EE*128,256),256,0,stream>>>(out_edge, wp1, bp1, ehp);
  wmma_gemm<ACT_NONE,true,false><<<dim3(2,8),256,0,stream>>>
      (NCh, wp2p, nullptr, P,512, nullptr,0, NN,1024,512);
  node_proj<<<CEILDIV(NN*4,256),256,0,stream>>>(NCf, bp2, rootw, rootb, xb, rtm);
  edge_msg<<<CEILDIV(EE,256),256,0,stream>>>(ehp, P, xb, msg);
  aggregate<<<NN,256,0,stream>>>(msg, rtm, out_actions);
}
